// RecurrentGemmaRecurrentBlock_26379689132850
// MI455X (gfx1250) — matrix operand order repaired and checked by reading, not compiled, not run on hardware
//
#include <hip/hip_runtime.h>

// ---------------------------------------------------------------------------
// RecurrentGemma recurrent block on MI455X (gfx1250, wave32, WMMA + TDM)
// fp32 math emulated via split-bf16 (hi + lo residual) on v_wmma_f32_16x16x32_bf16
// B tiles staged into LDS by the Tensor Data Mover (tensor_load_to_lds),
// double-buffered and overlapped with WMMA compute.
// ---------------------------------------------------------------------------

typedef __bf16 bf16;
typedef __attribute__((ext_vector_type(16))) __bf16 v16bf;
typedef __attribute__((ext_vector_type(8)))  __bf16 v8bf;
typedef __attribute__((ext_vector_type(8)))  float  v8f;
typedef __attribute__((ext_vector_type(4)))  unsigned int v4u;
typedef __attribute__((ext_vector_type(8)))  int v8i;
typedef __attribute__((ext_vector_type(4)))  int v4i;

#define HIDDEN 2560
#define LRU    2560
#define SEQ    4096
#define BATCH  2
#define HEADS  10
#define BW     256           // LRU / HEADS
#define CONVW  4
#define CS     128           // scan chunk length
#define NCH    (SEQ / CS)    // 32 chunks

union AFrag { v16bf v; v8bf h[2]; };

// ---------------------------------------------------------------------------
// fp32 -> (bf16 hi, bf16 lo) split
// ---------------------------------------------------------------------------
__global__ void split_f32_bf16(const float* __restrict__ x,
                               bf16* __restrict__ hi, bf16* __restrict__ lo,
                               long long n) {
  long long i = (long long)blockIdx.x * blockDim.x + threadIdx.x;
  if (i >= n) return;
  float v = x[i];
  bf16 h = (bf16)v;
  hi[i] = h;
  lo[i] = (bf16)(v - (float)h);
}

// ---------------------------------------------------------------------------
// TDM: DMA one [64 rows x 32 cols] bf16 tile (row stride = ldb elements) from
// global memory into LDS at byte offset ldsOff, packed row-major 64x32.
// D# layout per CDNA5 ISA (2-D tensor, groups 2/3 zero).
// 6-arg builtin form (clang-23 / therock-10.0 toolchain).
// ---------------------------------------------------------------------------
__device__ __forceinline__ void tdm_load_tile_64x32(unsigned ldsOff,
                                                    const bf16* gptr,
                                                    int ldb, int colsRemain) {
  unsigned long long ga = (unsigned long long)(size_t)gptr;
  v4u g0;
  g0.x = 1u;                                        // count=1 valid descriptor
  g0.y = ldsOff;                                    // lds_addr
  g0.z = (unsigned)(ga & 0xffffffffu);              // global_addr[31:0]
  g0.w = (unsigned)((ga >> 32) & 0x01ffffffu)       // global_addr[56:32]
         | (2u << 30);                              // type=2 ("image")
  v8i g1;
  g1[0] = (int)(1u << 16);                          // wg_mask=0, data_size=1 (2B)
  g1[1] = (int)(((unsigned)colsRemain & 0xffffu) << 16);          // dim0[15:0]
  g1[2] = (int)((((unsigned)colsRemain >> 16) & 0xffffu)          // dim0[31:16]
                | (64u << 16));                                   // dim1[15:0]=64
  g1[3] = (int)(32u << 16);                         // dim1[31:16]=0, tile_dim0=32
  g1[4] = (int)64u;                                 // tile_dim1=64, tile_dim2=0
  g1[5] = (int)(unsigned)ldb;                       // tensor_dim0_stride[31:0]
  g1[6] = 0;                                        // stride0[47:32], stride1[15:0]
  g1[7] = 0;                                        // stride1[47:16]
  const v4i zz4 = {0, 0, 0, 0};
  const v8i zz8 = {0, 0, 0, 0, 0, 0, 0, 0};
  __builtin_amdgcn_tensor_load_to_lds(g0, g1, zz4, zz4, zz8, 0);
}

// ---------------------------------------------------------------------------
// Split-bf16 WMMA GEMM:  C[m,n] = sum_k A[m,k]*B[n,k] + bias[n]  (+ optional gelu)
// A: [M,K] K-major (hi/lo), B: [N,K] K-major (hi/lo)  -> exactly X @ W^T
// Block: 256 threads = 8 waves. Tile: 256(M) x 64(N). Wave: 32(M) x 64(N).
// B tiles DMA'd into double-buffered LDS by TDM (wave 0), overlapped with WMMA.
// blockIdx.z batches block-diagonal heads via element strides aZoff/bZoff/cZoff.
// ---------------------------------------------------------------------------
__global__ __launch_bounds__(256) void gemm_split_bf16(
    const bf16* __restrict__ Ahi, const bf16* __restrict__ Alo, int lda, int aZoff,
    const bf16* __restrict__ Bhi, const bf16* __restrict__ Blo, int ldb, int bZoff,
    float* __restrict__ C, int ldc, int cZoff,
    const float* __restrict__ bias, int K, int act)
{
  __shared__ __align__(32) bf16 sBhi[2][64 * 32];
  __shared__ __align__(32) bf16 sBlo[2][64 * 32];

  const int z = blockIdx.z;
  Ahi += (size_t)z * aZoff;  Alo += (size_t)z * aZoff;
  Bhi += (size_t)z * bZoff;  Blo += (size_t)z * bZoff;
  C   += (size_t)z * cZoff;

  const int tid  = threadIdx.x;
  const int wave = tid >> 5;
  const int lane = tid & 31;
  const int m0 = blockIdx.x * 256;
  const int n0 = blockIdx.y * 64;

  // A fragment addressing (contiguous k order, same as the B read below):
  // lanes 0-15: row=lane, K 0..15; lanes 16-31: row=lane-16, K 16..31
  const int aRow = m0 + wave * 32 + (lane & 15);
  const int aK0  = (lane >> 4) << 4;                 // 0 or 16
  const bf16* aHiP0 = Ahi + (size_t)aRow * lda + aK0;
  const bf16* aLoP0 = Alo + (size_t)aRow * lda + aK0;
  const bf16* aHiP1 = aHiP0 + (size_t)16 * lda;      // second 16-row group
  const bf16* aLoP1 = aLoP0 + (size_t)16 * lda;

  // B fragment read: lanes 0-15: col=lane, K0..15; lanes 16-31: col=lane-16, K16..31
  const int fOff = (lane & 15) * 32 + ((lane >> 4) << 4);

  // LDS byte offsets of the two staging buffers (addr[31:0] of the LDS aperture)
  const unsigned offHi0 = (unsigned)(size_t)(void*)&sBhi[0][0];
  const unsigned offHi1 = (unsigned)(size_t)(void*)&sBhi[1][0];
  const unsigned offLo0 = (unsigned)(size_t)(void*)&sBlo[0][0];
  const unsigned offLo1 = (unsigned)(size_t)(void*)&sBlo[1][0];

  const v8f vzero = {0.f, 0.f, 0.f, 0.f, 0.f, 0.f, 0.f, 0.f};
  v8f acc[8];                                        // [mi*4 + nf]
#pragma unroll
  for (int i = 0; i < 8; ++i) acc[i] = vzero;

  const bf16* bHiTile = Bhi + (size_t)n0 * ldb;
  const bf16* bLoTile = Blo + (size_t)n0 * ldb;
  const int nStages = K >> 5;

  if (wave == 0) {                                   // prologue DMA: stage 0
    tdm_load_tile_64x32(offHi0, bHiTile, ldb, K);
    tdm_load_tile_64x32(offLo0, bLoTile, ldb, K);
  }

  for (int st = 0; st < nStages; ++st) {
    const int kc = st << 5;
    if (wave == 0) __builtin_amdgcn_s_wait_tensorcnt(0);  // stage st landed
    __syncthreads();
    if (wave == 0 && st + 1 < nStages) {             // DMA stage st+1, overlapped
      const int kn = kc + 32;
      const unsigned oh = ((st + 1) & 1) ? offHi1 : offHi0;
      const unsigned ol = ((st + 1) & 1) ? offLo1 : offLo0;
      tdm_load_tile_64x32(oh, bHiTile + kn, ldb, K - kn);
      tdm_load_tile_64x32(ol, bLoTile + kn, ldb, K - kn);
    }
    const bf16* sh = sBhi[st & 1];
    const bf16* sl = sBlo[st & 1];

    AFrag ahi0, alo0, ahi1, alo1;
    ahi0.h[0] = *(const v8bf*)(aHiP0 + kc);
    ahi0.h[1] = *(const v8bf*)(aHiP0 + kc + 8);
    alo0.h[0] = *(const v8bf*)(aLoP0 + kc);
    alo0.h[1] = *(const v8bf*)(aLoP0 + kc + 8);
    ahi1.h[0] = *(const v8bf*)(aHiP1 + kc);
    ahi1.h[1] = *(const v8bf*)(aHiP1 + kc + 8);
    alo1.h[0] = *(const v8bf*)(aLoP1 + kc);
    alo1.h[1] = *(const v8bf*)(aLoP1 + kc + 8);

#pragma unroll
    for (int nf = 0; nf < 4; ++nf) {
      v16bf bhi = *(const v16bf*)&sh[nf * 16 * 32 + fOff];
      v16bf blo = *(const v16bf*)&sl[nf * 16 * 32 + fOff];
      acc[nf] = __builtin_amdgcn_wmma_f32_16x16x32_bf16(false, ahi0.v, false, bhi,
                                                        (short)0, acc[nf], false, false);
      acc[nf] = __builtin_amdgcn_wmma_f32_16x16x32_bf16(false, ahi0.v, false, blo,
                                                        (short)0, acc[nf], false, false);
      acc[nf] = __builtin_amdgcn_wmma_f32_16x16x32_bf16(false, alo0.v, false, bhi,
                                                        (short)0, acc[nf], false, false);
      acc[4 + nf] = __builtin_amdgcn_wmma_f32_16x16x32_bf16(false, ahi1.v, false, bhi,
                                                        (short)0, acc[4 + nf], false, false);
      acc[4 + nf] = __builtin_amdgcn_wmma_f32_16x16x32_bf16(false, ahi1.v, false, blo,
                                                        (short)0, acc[4 + nf], false, false);
      acc[4 + nf] = __builtin_amdgcn_wmma_f32_16x16x32_bf16(false, alo1.v, false, bhi,
                                                        (short)0, acc[4 + nf], false, false);
    }
    __syncthreads();   // reads of buf[st&1] done before TDM overwrites it (st+2)
  }

  // C layout: VGPR r -> M = r (lanes 0-15) / r+8 (lanes 16-31); N = lane&15
#pragma unroll
  for (int mi = 0; mi < 2; ++mi) {
    const int mBase = m0 + wave * 32 + mi * 16 + ((lane >> 4) << 3);
#pragma unroll
    for (int nf = 0; nf < 4; ++nf) {
      const int n = n0 + nf * 16 + (lane & 15);
      const float bv = bias[n];
#pragma unroll
      for (int r = 0; r < 8; ++r) {
        float v = acc[mi * 4 + nf][r] + bv;
        if (act == 1) {  // gelu (tanh approx)
          float x3 = v * v * v;
          v = 0.5f * v * (1.f + tanhf(0.7978845608028654f * (v + 0.044715f * x3)));
        }
        C[(size_t)(mBase + r) * ldc + n] = v;
      }
    }
  }
}

// ---------------------------------------------------------------------------
// depthwise causal conv (width 4) + bf16 split of the output
// ---------------------------------------------------------------------------
__global__ void conv_split_kernel(const float* __restrict__ xb,
                                  const float* __restrict__ cw,
                                  const float* __restrict__ cb,
                                  float* __restrict__ co,
                                  bf16* __restrict__ chi, bf16* __restrict__ clo,
                                  long long n) {
  long long i = (long long)blockIdx.x * blockDim.x + threadIdx.x;
  if (i >= n) return;
  int c = (int)(i % LRU);
  long long bs = i / LRU;
  int s = (int)(bs % SEQ);
  long long b = bs / SEQ;
  float acc = cb[c];
#pragma unroll
  for (int k = 0; k < CONVW; ++k) {
    int sp = s + k - (CONVW - 1);
    if (sp >= 0) acc += xb[((b * SEQ + sp) * LRU) + c] * cw[c * CONVW + k];
  }
  co[i] = acc;
  bf16 h = (bf16)acc;
  chi[i] = h;
  clo[i] = (bf16)(acc - (float)h);
}

// ---------------------------------------------------------------------------
// RG-LRU elementwise prep: sigmoid gates, decay a, normalized x.
// In-place: gxbuf <- a, gabuf <- x_norm.
// ---------------------------------------------------------------------------
__global__ void rglru_prep(float* __restrict__ gxbuf, float* __restrict__ gabuf,
                           const float* __restrict__ conv_out,
                           const float* __restrict__ a_param,
                           const long long* __restrict__ pos,
                           long long n) {
  long long i = (long long)blockIdx.x * blockDim.x + threadIdx.x;
  if (i >= n) return;
  int c = (int)(i % LRU);
  long long bs = i / LRU;
  int s = (int)(bs % SEQ);
  int b = (int)(bs / SEQ);

  float gxl = gxbuf[i];
  float gal = gabuf[i];
  float gate_x = 1.f / (1.f + expf(-gxl));
  float gate_a = 1.f / (1.f + expf(-gal));
  float ap = a_param[c];
  float sp = (ap > 20.f) ? ap : log1pf(expf(ap));   // softplus
  float log_a = -8.f * gate_a * sp;
  float a = expf(log_a);
  float asq = expf(2.f * log_a);
  float mult = sqrtf(fmaxf(1.f - asq, 0.f));
  float gx = conv_out[i] * gate_x;
  bool reset = (pos[(long long)b * SEQ + s] == 0);
  if (reset) { mult = 1.f; a = 0.f; }
  gxbuf[i] = a;          // decay buffer
  gabuf[i] = gx * mult;  // normalized input buffer
}

// ---------------------------------------------------------------------------
// chunked linear-recurrence scan: h_t = a_t * h_{t-1} + x_t
// ---------------------------------------------------------------------------
__global__ void scan_carries(const float* __restrict__ a, const float* __restrict__ xn,
                             float* __restrict__ cA, float* __restrict__ cQ) {
  int bc = blockIdx.x;
  int b = bc / NCH, ch = bc % NCH;
  int c = blockIdx.y * blockDim.x + threadIdx.x;
  size_t base = ((size_t)b * SEQ + (size_t)ch * CS) * LRU + c;
  float A = 1.f, Q = 0.f;
  for (int t = 0; t < CS; ++t) {
    float at = a[base + (size_t)t * LRU];
    float xt = xn[base + (size_t)t * LRU];
    Q = at * Q + xt;
    A *= at;
  }
  size_t ci = ((size_t)b * NCH + ch) * LRU + c;
  cA[ci] = A;
  cQ[ci] = Q;
}

__global__ void scan_chunks(const float* __restrict__ cA, const float* __restrict__ cQ,
                            const float* __restrict__ h0, float* __restrict__ hstart) {
  int b = blockIdx.x;
  int c = blockIdx.y * blockDim.x + threadIdx.x;
  float h = h0[(size_t)b * LRU + c];
  for (int ch = 0; ch < NCH; ++ch) {
    size_t ci = ((size_t)b * NCH + ch) * LRU + c;
    hstart[ci] = h;
    h = cA[ci] * h + cQ[ci];
  }
}

// fixup pass fused with h * y_branch and bf16 split of the product
__global__ void scan_apply(const float* __restrict__ a, const float* __restrict__ xn,
                           const float* __restrict__ hstart, const float* __restrict__ yb,
                           bf16* __restrict__ phi, bf16* __restrict__ plo) {
  int bc = blockIdx.x;
  int b = bc / NCH, ch = bc % NCH;
  int c = blockIdx.y * blockDim.x + threadIdx.x;
  size_t base = ((size_t)b * SEQ + (size_t)ch * CS) * LRU + c;
  float h = hstart[((size_t)b * NCH + ch) * LRU + c];
  for (int t = 0; t < CS; ++t) {
    size_t idx = base + (size_t)t * LRU;
    h = a[idx] * h + xn[idx];
    float p = h * yb[idx];
    bf16 ph = (bf16)p;
    phi[idx] = ph;
    plo[idx] = (bf16)(p - (float)ph);
  }
}

// ---------------------------------------------------------------------------
// launch
// ---------------------------------------------------------------------------
extern "C" void kernel_launch(void* const* d_in, const int* in_sizes, int n_in,
                              void* d_out, int out_size, void* d_ws, size_t ws_size,
                              hipStream_t stream) {
  const float*     X       = (const float*)d_in[0];
  const long long* pos     = (const long long*)d_in[1];
  const float*     prev_h  = (const float*)d_in[2];
  const float*     w_y     = (const float*)d_in[3];
  const float*     b_y     = (const float*)d_in[4];
  const float*     w_x     = (const float*)d_in[5];
  const float*     b_x     = (const float*)d_in[6];
  const float*     w_out   = (const float*)d_in[7];
  const float*     b_out   = (const float*)d_in[8];
  const float*     conv_w  = (const float*)d_in[9];
  const float*     conv_b  = (const float*)d_in[10];
  const float*     a_param = (const float*)d_in[11];
  const float*     ig_w    = (const float*)d_in[12];
  const float*     ig_b    = (const float*)d_in[13];
  const float*     ag_w    = (const float*)d_in[14];
  const float*     ag_b    = (const float*)d_in[15];
  float* out = (float*)d_out;

  const long long MTOK = (long long)BATCH * SEQ;      // 8192 tokens
  const long long NX   = MTOK * HIDDEN;               // 20,971,520 elements
  const long long NW   = (long long)HIDDEN * HIDDEN;  // 2560*2560
  const long long NG   = (long long)BW * LRU;         // 256*2560

  char* ws = (char*)d_ws;
  size_t off = 0;
  auto alloc = [&](size_t bytes) -> char* {
    char* p = ws + off;
    off = (off + bytes + 255) & ~(size_t)255;
    return p;
  };

  bf16* XHI  = (bf16*)alloc(NX * 2);   // NX*2 is 256-aligned -> XHI..XLO contiguous
  bf16* XLO  = (bf16*)alloc(NX * 2);
  bf16* WYHI = (bf16*)alloc(NW * 2);
  bf16* WYLO = (bf16*)alloc(NW * 2);
  bf16* WXHI = (bf16*)alloc(NW * 2);
  bf16* WXLO = (bf16*)alloc(NW * 2);
  bf16* WOHI = (bf16*)alloc(NW * 2);
  bf16* WOLO = (bf16*)alloc(NW * 2);
  bf16* IGHI = (bf16*)alloc(NG * 2);
  bf16* IGLO = (bf16*)alloc(NG * 2);
  bf16* AGHI = (bf16*)alloc(NG * 2);
  bf16* AGLO = (bf16*)alloc(NG * 2);
  float* YB   = (float*)alloc(NX * 4);
  float* XB   = (float*)alloc(NX * 4);
  float* CONV = (float*)alloc(NX * 4);
  bf16* CHI  = (bf16*)alloc(NX * 2);
  bf16* CLO  = (bf16*)alloc(NX * 2);
  float* cA  = (float*)alloc((size_t)BATCH * NCH * LRU * 4);
  float* cQ  = (float*)alloc((size_t)BATCH * NCH * LRU * 4);
  float* HST = (float*)alloc((size_t)BATCH * NCH * LRU * 4);

  // buffer reuse (strictly sequential stream):
  float* GX = XB;          // gate_x linear -> later decay a
  float* GA = (float*)XHI; // gate_a linear (XHI+XLO region dead after GEMM 1/2) -> later x_norm
  bf16* PHI = CHI;         // h * y_branch split reuses conv-split region
  bf16* PLO = CLO;

  const int TB = 256;
  auto blocks = [](long long n) { return (unsigned)((n + 255) / 256); };

  // 1) fp32 -> bf16 hi/lo splits
  split_f32_bf16<<<blocks(NX), TB, 0, stream>>>(X, XHI, XLO, NX);
  split_f32_bf16<<<blocks(NW), TB, 0, stream>>>(w_y, WYHI, WYLO, NW);
  split_f32_bf16<<<blocks(NW), TB, 0, stream>>>(w_x, WXHI, WXLO, NW);
  split_f32_bf16<<<blocks(NW), TB, 0, stream>>>(w_out, WOHI, WOLO, NW);
  split_f32_bf16<<<blocks(NG), TB, 0, stream>>>(ig_w, IGHI, IGLO, NG);
  split_f32_bf16<<<blocks(NG), TB, 0, stream>>>(ag_w, AGHI, AGLO, NG);

  // 2) y_branch = gelu(X @ w_y^T + b_y);  x_branch = X @ w_x^T + b_x
  dim3 gBig((unsigned)(MTOK / 256), LRU / 64, 1);
  gemm_split_bf16<<<gBig, TB, 0, stream>>>(XHI, XLO, HIDDEN, 0,
                                           WYHI, WYLO, HIDDEN, 0,
                                           YB, LRU, 0, b_y, HIDDEN, 1);
  gemm_split_bf16<<<gBig, TB, 0, stream>>>(XHI, XLO, HIDDEN, 0,
                                           WXHI, WXLO, HIDDEN, 0,
                                           XB, LRU, 0, b_x, HIDDEN, 0);

  // 3) causal depthwise conv + split
  conv_split_kernel<<<blocks(NX), TB, 0, stream>>>(XB, conv_w, conv_b, CONV, CHI, CLO, NX);

  // 4) block-diagonal gate GEMMs (blockIdx.z = head)
  dim3 gGate((unsigned)(MTOK / 256), BW / 64, HEADS);
  gemm_split_bf16<<<gGate, TB, 0, stream>>>(CHI, CLO, LRU, BW,
                                            IGHI, IGLO, BW, BW * BW,
                                            GX, LRU, BW, ig_b, BW, 0);
  gemm_split_bf16<<<gGate, TB, 0, stream>>>(CHI, CLO, LRU, BW,
                                            AGHI, AGLO, BW, BW * BW,
                                            GA, LRU, BW, ag_b, BW, 0);

  // 5) gates -> (a, x_norm) in place
  rglru_prep<<<blocks(NX), TB, 0, stream>>>(GX, GA, CONV, a_param, pos, NX);

  // 6) chunked linear scan, fixup fused with h*y_branch and bf16 split
  dim3 gScan(BATCH * NCH, LRU / TB);
  scan_carries<<<gScan, TB, 0, stream>>>(GX, GA, cA, cQ);
  scan_chunks<<<dim3(BATCH, LRU / TB), TB, 0, stream>>>(cA, cQ, prev_h, HST);
  scan_apply<<<gScan, TB, 0, stream>>>(GX, GA, HST, YB, PHI, PLO);

  // 7) out = (h * y_branch) @ w_out^T + b_out
  dim3 gOut((unsigned)(MTOK / 256), HIDDEN / 64, 1);
  gemm_split_bf16<<<gOut, TB, 0, stream>>>(PHI, PLO, LRU, 0,
                                           WOHI, WOLO, LRU, 0,
                                           out, HIDDEN, 0, b_out, LRU, 0);
}
